// SelfAttention_39522289058172
// MI455X (gfx1250) — matrix operand order repaired and checked by reading, not compiled, not run on hardware
//
#include <hip/hip_runtime.h>

typedef __attribute__((ext_vector_type(16))) _Float16     v16h;
typedef __attribute__((ext_vector_type(8)))  float        v8f;
typedef __attribute__((ext_vector_type(4)))  unsigned int v4u;
typedef __attribute__((ext_vector_type(4)))  int          v4i;

union FragH { v4u u[2]; v16h h; };

// ---- CDNA5 async global->LDS (ASYNCcnt) with compile-safe fallback ----
#if defined(__HIP_DEVICE_COMPILE__) && \
    __has_builtin(__builtin_amdgcn_global_load_async_to_lds_b128)
#define USE_ASYNC_LDS 1
#else
#define USE_ASYNC_LDS 0
#endif

#if USE_ASYNC_LDS
typedef __attribute__((address_space(1))) v4i* gas_v4i_p;  // global
typedef __attribute__((address_space(3))) v4i* las_v4i_p;  // LDS
#endif

__device__ __forceinline__ void cp16_g2l(const _Float16* g, _Float16* l) {
#if USE_ASYNC_LDS
  __builtin_amdgcn_global_load_async_to_lds_b128((gas_v4i_p)g, (las_v4i_p)l,
                                                 0, 0);
#else
  *(v4u*)l = *(const v4u*)g;  // sync fallback (DScnt handled by barrier)
#endif
}

__device__ __forceinline__ void wait_async0() {
#if USE_ASYNC_LDS
#if __has_builtin(__builtin_amdgcn_s_wait_asynccnt)
  __builtin_amdgcn_s_wait_asynccnt(0);
#else
  asm volatile("s_wait_asynccnt 0x0" ::: "memory");
#endif
#endif
}

// ---------------- f32 -> f16 conversion ----------------
__global__ __launch_bounds__(256) void cvt_f32_f16(const float* __restrict__ s,
                                                   _Float16* __restrict__ d, int n) {
  int i  = blockIdx.x * 256 + threadIdx.x;
  int st = gridDim.x * 256;
  for (; i < n; i += st) d[i] = (_Float16)s[i];
}

// -------- f32 [R x Cc] -> f16 transposed [Cc x R], 64x64 LDS tiles --------
__global__ __launch_bounds__(256) void tcvt_f32_f16t(const float* __restrict__ s,
                                                     _Float16* __restrict__ d,
                                                     int R, int Cc) {
  __shared__ _Float16 t[64][65];
  const int tid = threadIdx.x;
  const int r0 = blockIdx.y * 64, c0 = blockIdx.x * 64;
  const int tr = tid >> 6;   // 0..3
  const int tc = tid & 63;   // 0..63
#pragma unroll
  for (int i = 0; i < 16; ++i) {
    int r = tr * 16 + i;
    t[r][tc] = (_Float16)s[(size_t)(r0 + r) * Cc + c0 + tc];
  }
  __syncthreads();
#pragma unroll
  for (int i = 0; i < 16; ++i) {
    int x = tr * 16 + i;
    d[(size_t)(c0 + x) * R + r0 + tc] = t[tc][x];
  }
}

// ---------------- WMMA GEMM (B always stored N x K row-major) ----------------
// C[m,n] = alpha * sum_k A[m,k] * B[n,k]
// CAUSAL  : fuse causal mask (-inf for col>m), skip blocks above diagonal.
// CAUSAL_A: A (=P) is lower-triangular -> clip K loop at m0+BM.
// OUT_TRANS: write D transposed per batch-of-2048-rows: dst[b][col][t]
//            (used to produce V^T directly from the V projection).
// Block 256x128, K-step 32, 8 waves, wave = 4x4 WMMA tiles. Double-buffered
// LDS fed by async global->LDS copies; ONE barrier per K-step.
template <bool OUT_F16, bool CAUSAL, bool CAUSAL_A, bool OUT_TRANS>
__global__ __launch_bounds__(256) void gemm_wmma(
    const _Float16* __restrict__ A, const _Float16* __restrict__ B,
    void* __restrict__ Cv, int K, int lda, int ldb, int ldc,
    long long sA, long long sB, long long sC, float alpha) {
  constexpr int BM = 256;
  __shared__ alignas(16) _Float16 As[2][256][40];  // [m][k], 80B row pitch
  __shared__ alignas(16) _Float16 Bs[2][128][40];  // [n][k]

  const int m0 = blockIdx.y * BM;
  const int n0 = blockIdx.x * 128;
  if (CAUSAL && n0 >= m0 + BM) return;  // fully masked: never read downstream

  const int tid  = threadIdx.x;
  const int wave = tid >> 5;
  const int lane = tid & 31;
  const int wr   = wave & 3;   // 4 row strips of 64
  const int wc   = wave >> 2;  // 2 col strips of 64
  const int lr   = lane & 15;
  const int hi   = lane >> 4;

  const _Float16* Ab = A + (size_t)blockIdx.z * sA;
  const _Float16* Bb = B + (size_t)blockIdx.z * sB;

  const int Kend = CAUSAL_A ? ((m0 + BM < K) ? (m0 + BM) : K) : K;

  // per-thread staging coordinates (16B chunks)
  const int arow = tid >> 2, akc = (tid & 3) << 3;  // + p*64 rows
  const int brow = tid >> 2, bkc = (tid & 3) << 3;  // + p*64 rows (p<2)

  auto stage = [&](int kt, int buf) {
#pragma unroll
    for (int p = 0; p < 4; ++p)
      cp16_g2l(Ab + (size_t)(m0 + arow + p * 64) * lda + kt + akc,
               &As[buf][arow + p * 64][akc]);
#pragma unroll
    for (int p = 0; p < 2; ++p)
      cp16_g2l(Bb + (size_t)(n0 + brow + p * 64) * ldb + kt + bkc,
               &Bs[buf][brow + p * 64][bkc]);
  };

  v8f acc[4][4] = {};

  stage(0, 0);  // prologue
  int buf = 0;
  for (int kt = 0; kt < Kend; kt += 32, buf ^= 1) {
    wait_async0();    // my copies into `buf` have landed in LDS
    __syncthreads();  // everyone's copies landed; prior compute finished
    if (kt + 32 < Kend) stage(kt + 32, buf ^ 1);  // overlaps WMMAs below

    // A 16x32 frag: lane<16 -> {K0..7,K16..23}; lane>=16 -> {K8..15,K24..31}
    v16h afr[4];
#pragma unroll
    for (int i = 0; i < 4; ++i) {
      int r = wr * 64 + i * 16 + lr;
      FragH f;
      f.u[0] = *(const v4u*)&As[buf][r][hi * 8];
      f.u[1] = *(const v4u*)&As[buf][r][hi * 8 + 16];
      afr[i] = f.h;
    }
    // B 32x16 frag: lane<16 -> {K0..7,K16..23}; lane>=16 -> {K8..15,K24..31}
#pragma unroll
    for (int j = 0; j < 4; ++j) {
      int c = wc * 64 + j * 16 + lr;
      FragH f;
      f.u[0] = *(const v4u*)&Bs[buf][c][hi * 8];
      f.u[1] = *(const v4u*)&Bs[buf][c][hi * 8 + 16];
      v16h bfr = f.h;
#pragma unroll
      for (int i = 0; i < 4; ++i)
        acc[i][j] = __builtin_amdgcn_wmma_f32_16x16x32_f16(
            false, afr[i], false, bfr, (short)0, acc[i][j], false, false);
    }
  }

  // ---- epilogue: D layout VGPR r -> M = r + 8*hi, N = lane&15 ----
#pragma unroll
  for (int i = 0; i < 4; ++i) {
#pragma unroll
    for (int j = 0; j < 4; ++j) {
      int mb  = m0 + wr * 64 + i * 16 + hi * 8;
      int col = n0 + wc * 64 + j * 16 + lr;
#pragma unroll
      for (int r = 0; r < 8; ++r) {
        int   m = mb + r;
        float v = acc[i][j][r] * alpha;
        if (CAUSAL && col > m) v = -__builtin_inff();
        if (OUT_TRANS) {  // dst[b][col][t], b = m/2048, t = m%2048
          _Float16* Cb = (_Float16*)Cv;
          Cb[(size_t)(m >> 11) * sC + (size_t)col * ldc + (m & 2047)] =
              (_Float16)v;
        } else if (OUT_F16) {
          _Float16* Cb = (_Float16*)Cv + (size_t)blockIdx.z * sC;
          Cb[(size_t)m * ldc + col] = (_Float16)v;
        } else {
          float* Cb = (float*)Cv + (size_t)blockIdx.z * sC;
          Cb[(size_t)m * ldc + col] = v;
        }
      }
    }
  }
}

// ---------------- causal row softmax ----------------
// Reads/writes only s < blockEnd = (t/256+1)*256 (region written by the
// causal GEMM). P (f16) aliases S (f32) in place: both 8192B row pitch.
__global__ __launch_bounds__(256) void softmax_causal(const float* __restrict__ S,
                                                      _Float16* __restrict__ P) {
  __shared__ float red[256];
  const int    tid = threadIdx.x;
  const size_t r   = blockIdx.x;
  const int    t   = (int)(r & 2047);
  const int    blockEnd = ((t >> 8) + 1) << 8;
  const float*    row  = S + r * 2048;
  _Float16*       prow = P + r * 4096;
  const int  c0  = tid * 8;
  const bool act = c0 < blockEnd;

  float x[8];
#pragma unroll
  for (int i = 0; i < 8; ++i) x[i] = -__builtin_inff();
  if (act) {
#pragma unroll
    for (int i = 0; i < 8; ++i) x[i] = row[c0 + i];
  }

  float m = x[0];
#pragma unroll
  for (int i = 1; i < 8; ++i) m = fmaxf(m, x[i]);
  red[tid] = m;
  __syncthreads();
  for (int s = 128; s > 0; s >>= 1) {
    if (tid < s) red[tid] = fmaxf(red[tid], red[tid + s]);
    __syncthreads();
  }
  m = red[0];
  __syncthreads();

  float e[8], sum = 0.f;
#pragma unroll
  for (int i = 0; i < 8; ++i) {
    e[i] = __expf(x[i] - m);
    sum += e[i];
  }
  red[tid] = sum;
  __syncthreads();
  for (int s = 128; s > 0; s >>= 1) {
    if (tid < s) red[tid] += red[tid + s];
    __syncthreads();
  }
  const float inv = 1.0f / red[0];
  if (act) {
#pragma unroll
    for (int i = 0; i < 8; ++i) prow[c0 + i] = (_Float16)(e[i] * inv);
  }
}

// ---------------- launcher ----------------
extern "C" void kernel_launch(void* const* d_in, const int* in_sizes, int n_in,
                              void* d_out, int out_size, void* d_ws,
                              size_t ws_size, hipStream_t stream) {
  (void)in_sizes; (void)n_in; (void)out_size; (void)ws_size;
  const int B = 4, T = 2048, C = 1024;
  const long long TC = (long long)T * C;

  const float* X  = (const float*)d_in[0];
  const float* Wq = (const float*)d_in[1];
  const float* Wk = (const float*)d_in[2];
  const float* Wv = (const float*)d_in[3];
  const float* Wo = (const float*)d_in[4];

  char* ws = (char*)d_ws;
  const size_t MB = 1ull << 20;
  _Float16* Xh   = (_Float16*)(ws + 0 * MB);   // 16MB (reused as Oh)
  _Float16* WqT  = (_Float16*)(ws + 16 * MB);  // 2MB each, f16 transposed
  _Float16* WkT  = (_Float16*)(ws + 18 * MB);
  _Float16* WvT  = (_Float16*)(ws + 20 * MB);
  _Float16* WoT  = (_Float16*)(ws + 22 * MB);
  _Float16* Qh   = (_Float16*)(ws + 24 * MB);  // 16MB
  _Float16* Kh   = (_Float16*)(ws + 40 * MB);  // 16MB
  _Float16* VT   = (_Float16*)(ws + 56 * MB);  // 16MB, [B][C][T]
  float*    Sf   = (float*)(ws + 72 * MB);     // 64MB scores f32
  _Float16* P    = (_Float16*)(ws + 72 * MB);  // f16 probs in place
  _Float16* Oh   = Xh;

  dim3 blk(256);

  // 0) conversions: X -> f16; weights -> f16 transposed (N x K)
  cvt_f32_f16<<<dim3(2048), blk, 0, stream>>>(X, Xh, B * T * C);
  dim3 gt(16, 16, 1);
  tcvt_f32_f16t<<<gt, blk, 0, stream>>>(Wq, WqT, C, C);
  tcvt_f32_f16t<<<gt, blk, 0, stream>>>(Wk, WkT, C, C);
  tcvt_f32_f16t<<<gt, blk, 0, stream>>>(Wv, WvT, C, C);
  tcvt_f32_f16t<<<gt, blk, 0, stream>>>(Wo, WoT, C, C);

  // 1) projections (B operands are N x K). V written transposed -> VT.
  dim3 gproj(C / 128, (B * T) / 256, 1);  // (8, 32)
  gemm_wmma<true, false, false, false><<<gproj, blk, 0, stream>>>(
      Xh, WqT, Qh, C, C, C, C, 0, 0, 0, 1.0f);
  gemm_wmma<true, false, false, false><<<gproj, blk, 0, stream>>>(
      Xh, WkT, Kh, C, C, C, C, 0, 0, 0, 1.0f);
  gemm_wmma<true, false, false, true><<<gproj, blk, 0, stream>>>(
      Xh, WvT, VT, C, C, C, T, 0, 0, TC, 1.0f);  // ldc=T, sC=TC (per batch)

  // 2) S = (Q @ K^T)/sqrt(C); B = Kh already N x K. Above-diagonal skipped.
  dim3 gs(T / 128, T / 256, B);  // (16, 8, 4)
  gemm_wmma<false, true, false, false><<<gs, blk, 0, stream>>>(
      Qh, Kh, Sf, C, C, C, T, TC, TC, (long long)T * T, 0.03125f);

  // 3) causal softmax -> P (f16, in place over S)
  softmax_causal<<<dim3(B * T), blk, 0, stream>>>(Sf, P);

  // 4) O = P @ V; B = VT (N=C x K=T per batch). K-loop causally clipped.
  dim3 go(C / 128, T / 256, B);  // (8, 8, 4)
  gemm_wmma<true, false, true, false><<<go, blk, 0, stream>>>(
      P, VT, Oh, T, 2 * T, T, C, (long long)T * 2 * T, TC, TC, 1.0f);

  // 5) out = O @ Wo (f32 to d_out); B = WoT
  gemm_wmma<false, false, false, false><<<gproj, blk, 0, stream>>>(
      Oh, WoT, d_out, C, C, C, C, 0, 0, 0, 1.0f);
}
